// RGCN_38929583571527
// MI455X (gfx1250) — compile-verified, matrix operand order checked by reading, not run on hardware
//
#include <hip/hip_runtime.h>
#include <hip/hip_fp16.h>

typedef __attribute__((ext_vector_type(16))) _Float16 v16h;
typedef __attribute__((ext_vector_type(8)))  _Float16 v8h;
typedef __attribute__((ext_vector_type(8)))  float    v8f;

#define NN   50000
#define RR   8
#define HH   128
#define LL   32
#define EE   1600000
#define CEXT 288   // 256 cols (R*L) for xw + 32 cols for x@root2

// ---------------------------------------------------------------------------
// Pass 1a: per-(node,relation) edge counts
// ---------------------------------------------------------------------------
__global__ void k_count(const int* __restrict__ dst, const int* __restrict__ et,
                        float* __restrict__ cnt, int E) {
  int i = blockIdx.x * blockDim.x + threadIdx.x;
  if (i < E) atomicAdd(&cnt[dst[i] * RR + et[i]], 1.0f);
}

// Pass 1b: invert counts in place (0 stays 0)
__global__ void k_inv(float* __restrict__ cnt, int n) {
  int i = blockIdx.x * blockDim.x + threadIdx.x;
  if (i < n) {
    float c = cnt[i];
    cnt[i] = (c > 0.0f) ? (1.0f / c) : 0.0f;
  }
}

// ---------------------------------------------------------------------------
// Layer 1: one wave per edge; 32 lanes x float4 = 128 floats = H.
// x[dst,:] += weight1[et,src,:] * inv_cnt[dst,et]
// ---------------------------------------------------------------------------
__global__ void __launch_bounds__(256)
k_l1_scatter(const float* __restrict__ w1, const int* __restrict__ src,
             const int* __restrict__ dst, const int* __restrict__ et,
             const float* __restrict__ inv, float* __restrict__ x, int E) {
  int e = blockIdx.x * 8 + (threadIdx.x >> 5);
  if (e >= E) return;
  int lane = threadIdx.x & 31;
  int s = src[e], d = dst[e], r = et[e];
  float sc = inv[d * RR + r];
  const float4* row = (const float4*)(w1 + ((size_t)r * NN + s) * HH);
  float4 v = row[lane];
  float* xp = x + (size_t)d * HH + lane * 4;
  atomicAdd(xp + 0, v.x * sc);
  atomicAdd(xp + 1, v.y * sc);
  atomicAdd(xp + 2, v.z * sc);
  atomicAdd(xp + 3, v.w * sc);
}

// x = relu(x + root1 + bias1); emit f16 copy for WMMA
__global__ void k_finish_x(const float* __restrict__ x, const float* __restrict__ root1,
                           const float* __restrict__ bias1, _Float16* __restrict__ xh, int n) {
  int i = blockIdx.x * blockDim.x + threadIdx.x;
  if (i < n) {
    float v = x[i] + root1[i] + bias1[i & (HH - 1)];
    v = v > 0.0f ? v : 0.0f;
    xh[i] = (_Float16)v;
  }
}

// ---------------------------------------------------------------------------
// Pack B = [weight2 reshaped to H x 256 | root2 (H x 32)] as f16, K-major:
// b2t[c*HH + h], so B fragments load as two contiguous 16B chunks per lane.
// ---------------------------------------------------------------------------
__global__ void k_pack_b(const float* __restrict__ w2, const float* __restrict__ root2,
                         _Float16* __restrict__ b2t) {
  int i = blockIdx.x * blockDim.x + threadIdx.x;
  if (i >= CEXT * HH) return;
  int c = i / HH, h = i - c * HH;
  float v;
  if (c < RR * LL) {
    int r = c / LL, l = c - r * LL;
    v = w2[((size_t)r * HH + h) * LL + l];
  } else {
    v = root2[h * LL + (c - RR * LL)];
  }
  b2t[i] = (_Float16)v;
}

// ---------------------------------------------------------------------------
// GEMM: xw[N x 288] = xh[N x 128] @ b2t^T, f16 WMMA with f32 accumulation.
// Block = 256 threads (8 waves) handles a 16-row strip; A strip staged in LDS.
// 16-bit A/B fragment layout per ISA: lane&15 = row/col, (lane>>4)*8 = K base.
// ---------------------------------------------------------------------------
__global__ void __launch_bounds__(256)
k_gemm(const _Float16* __restrict__ xh, const _Float16* __restrict__ b2t,
       float* __restrict__ xw) {
  __shared__ __align__(16) _Float16 At[16 * HH];   // 4 KB
  int row0 = blockIdx.x * 16;                      // N = 3125*16, no guard needed
  {
    // 2048 halves, 256 threads -> one 16B load each
    int t = threadIdx.x;
    const float4* g = (const float4*)(xh + (size_t)row0 * HH) + t;
    ((float4*)At)[t] = *g;
  }
  __syncthreads();

  int lane = threadIdx.x & 31;
  int wv   = threadIdx.x >> 5;
  int rl   = lane & 15;          // row (A) / col (B) within tile
  int g8   = (lane >> 4) * 8;    // K sub-block select; also C-row offset

  for (int tile = wv; tile < CEXT / 16; tile += 8) {
    int col0 = tile * 16;
    v8f c = {};
#pragma unroll
    for (int k0 = 0; k0 < HH; k0 += 32) {
      union { v16h v; v8h h[2]; } a, b;
      a.h[0] = *(const v8h*)(At + rl * HH + k0 + g8);
      a.h[1] = *(const v8h*)(At + rl * HH + k0 + 16 + g8);
      b.h[0] = *(const v8h*)(b2t + (size_t)(col0 + rl) * HH + k0 + g8);
      b.h[1] = *(const v8h*)(b2t + (size_t)(col0 + rl) * HH + k0 + 16 + g8);
      c = __builtin_amdgcn_wmma_f32_16x16x32_f16(false, a.v, false, b.v,
                                                 (short)0, c, false, false);
    }
#pragma unroll
    for (int v = 0; v < 8; ++v) {
      int rowm = v + g8;  // C layout: VGPR v -> M = v + 8*(lane>>4), N = lane&15
      xw[(size_t)(row0 + rowm) * CEXT + col0 + rl] = c[v];
    }
  }
}

// out init: x@root2 columns (256..287 of xw) + bias2
__global__ void k_init_out(const float* __restrict__ xw, const float* __restrict__ bias2,
                           float* __restrict__ out, int n) {
  int i = blockIdx.x * blockDim.x + threadIdx.x;
  if (i < n) {
    int node = i >> 5, l = i & 31;
    out[i] = xw[(size_t)node * CEXT + 256 + l] + bias2[l];
  }
}

// ---------------------------------------------------------------------------
// Layer 2: one wave per edge; lane = label.
// out[dst,:] += xw[src, et*32 .. et*32+31] * inv_cnt[dst,et]
// ---------------------------------------------------------------------------
__global__ void __launch_bounds__(256)
k_l2_scatter(const float* __restrict__ xw, const int* __restrict__ src,
             const int* __restrict__ dst, const int* __restrict__ et,
             const float* __restrict__ inv, float* __restrict__ out, int E) {
  int e = blockIdx.x * 8 + (threadIdx.x >> 5);
  if (e >= E) return;
  int lane = threadIdx.x & 31;
  int s = src[e], d = dst[e], r = et[e];
  float sc = inv[d * RR + r];
  float v = xw[(size_t)s * CEXT + r * LL + lane];
  atomicAdd(out + (size_t)d * LL + lane, v * sc);
}

__global__ void k_sigmoid(float* __restrict__ out, int n) {
  int i = blockIdx.x * blockDim.x + threadIdx.x;
  if (i < n) out[i] = 1.0f / (1.0f + __expf(-out[i]));
}

// ---------------------------------------------------------------------------
extern "C" void kernel_launch(void* const* d_in, const int* in_sizes, int n_in,
                              void* d_out, int out_size, void* d_ws, size_t ws_size,
                              hipStream_t stream) {
  const float* w1    = (const float*)d_in[0];
  const float* root1 = (const float*)d_in[1];
  const float* bias1 = (const float*)d_in[2];  (void)bias1;
  const float* w2    = (const float*)d_in[3];
  const float* root2 = (const float*)d_in[4];
  const float* bias2 = (const float*)d_in[5];
  const int*   eidx  = (const int*)d_in[6];
  const int*   etyp  = (const int*)d_in[7];
  const int* src = eidx;
  const int* dst = eidx + EE;
  float* out = (float*)d_out;

  char* ws = (char*)d_ws;
  size_t off = 0;
  auto carve = [&](size_t bytes) -> char* {
    char* p = ws + off;
    off = (off + bytes + 255) & ~(size_t)255;
    return p;
  };
  float*    cnt  = (float*)carve(sizeof(float) * NN * RR);               // 1.6 MB
  float*    xbuf = (float*)carve(sizeof(float) * (size_t)NN * HH);       // 25.6 MB
  _Float16* xh   = (_Float16*)carve(sizeof(_Float16) * (size_t)NN * HH); // 12.8 MB
  _Float16* b2t  = (_Float16*)carve(sizeof(_Float16) * CEXT * HH);       // 74 KB
  float*    xw   = (float*)carve(sizeof(float) * (size_t)NN * CEXT);     // 57.6 MB

  hipMemsetAsync(cnt,  0, sizeof(float) * NN * RR, stream);
  hipMemsetAsync(xbuf, 0, sizeof(float) * (size_t)NN * HH, stream);

  k_count     <<<(EE + 255) / 256, 256, 0, stream>>>(dst, etyp, cnt, EE);
  k_inv       <<<(NN * RR + 255) / 256, 256, 0, stream>>>(cnt, NN * RR);
  k_l1_scatter<<<(EE + 7) / 8, 256, 0, stream>>>(w1, src, dst, etyp, cnt, xbuf, EE);
  k_finish_x  <<<(NN * HH + 255) / 256, 256, 0, stream>>>(xbuf, root1, (const float*)d_in[2], xh, NN * HH);
  k_pack_b    <<<(CEXT * HH + 255) / 256, 256, 0, stream>>>(w2, root2, b2t);
  k_gemm      <<<NN / 16, 256, 0, stream>>>(xh, b2t, xw);
  k_init_out  <<<(NN * LL + 255) / 256, 256, 0, stream>>>(xw, bias2, out, NN * LL);
  k_l2_scatter<<<(EE + 7) / 8, 256, 0, stream>>>(xw, src, dst, etyp, cnt, out, EE);
  k_sigmoid   <<<(NN * LL + 255) / 256, 256, 0, stream>>>(out, NN * LL);
}
